// GCN_76184129896739
// MI455X (gfx1250) — compile-verified, matrix operand order checked by reading, not run on hardware
//
#include <hip/hip_runtime.h>
#include <math.h>

#define IN_F  256
#define HID_F 256
#define OUT_F 128

typedef __bf16 bf16_t;
typedef bf16_t v16bf __attribute__((ext_vector_type(16)));
typedef bf16_t v8bf  __attribute__((ext_vector_type(8)));
typedef float  v8f   __attribute__((ext_vector_type(8)));

__device__ __forceinline__ void atomAddF(float* p, float v) {
  __hip_atomic_fetch_add(p, v, __ATOMIC_RELAXED, __HIP_MEMORY_SCOPE_AGENT);
}

// ---------------- utility kernels ----------------

__global__ void fill_f32(float* __restrict__ p, float v, long long n) {
  long long i = (long long)blockIdx.x * blockDim.x + threadIdx.x;
  if (i < n) p[i] = v;
}

__global__ void deg_kernel(const long long* __restrict__ dst, long long E,
                           float* __restrict__ deg) {
  long long e = (long long)blockIdx.x * blockDim.x + threadIdx.x;
  if (e < E) atomAddF(&deg[dst[e]], 1.0f);
}

__global__ void rsqrt_kernel(float* __restrict__ d, long long n) {
  long long i = (long long)blockIdx.x * blockDim.x + threadIdx.x;
  if (i < n) {
    float v = d[i];
    d[i] = (v > 0.0f) ? rsqrtf(v) : 0.0f;
  }
}

__global__ void cvt_bf16_kernel(const float* __restrict__ in,
                                bf16_t* __restrict__ out, long long n) {
  long long i = (long long)blockIdx.x * blockDim.x + threadIdx.x;
  if (i < n) out[i] = (bf16_t)in[i];
}

// W is (K x Nout) row-major fp32 -> Wt is (Nout x K) row-major bf16
__global__ void transpose_cvt_kernel(const float* __restrict__ W,
                                     bf16_t* __restrict__ Wt, int K, int Nout) {
  int i = blockIdx.x * blockDim.x + threadIdx.x;
  if (i >= K * Nout) return;
  int k = i / Nout, n = i % Nout;
  Wt[(long long)n * K + k] = (bf16_t)W[i];
}

// ---------------- WMMA GEMM: C(N x Nout) = A(N x K) * Wt^T ----------------
// A: bf16 row-major (row, K contiguous). Wt: bf16 (outCol, K contiguous).
// One wave computes a 16x16 output tile; 4 waves/block cover 64 columns.

__global__ void __launch_bounds__(128)
gemm_wmma_bf16(const bf16_t* __restrict__ A, const bf16_t* __restrict__ Bt,
               float* __restrict__ C, int K, int Nout) {
  const int lane = threadIdx.x & 31;
  const int wave = threadIdx.x >> 5;
  const int rowTile = blockIdx.x;
  const int colTile = blockIdx.y * 4 + wave;
  const int r  = lane & 15;           // matrix row (A) / out-column row (B)
  const int kh = (lane >> 4) << 3;    // 0 for lanes 0-15, 8 for lanes 16-31
  const long long arow = (long long)(rowTile * 16 + r) * K;
  const long long brow = (long long)(colTile * 16 + r) * K;

  v8f acc = {};
  for (int kb = 0; kb < K; kb += 32) {
    const v8bf alo = *(const v8bf*)(A + arow + kb + kh);
    const v8bf ahi = *(const v8bf*)(A + arow + kb + 16 + kh);
    const v8bf blo = *(const v8bf*)(Bt + brow + kb + kh);
    const v8bf bhi = *(const v8bf*)(Bt + brow + kb + 16 + kh);
    v16bf a, b;
#pragma unroll
    for (int i = 0; i < 8; ++i) {
      a[i] = alo[i]; a[8 + i] = ahi[i];
      b[i] = blo[i]; b[8 + i] = bhi[i];
    }
    acc = __builtin_amdgcn_wmma_f32_16x16x32_bf16(
        /*neg_a=*/false, a, /*neg_b=*/false, b,
        /*c_mod=*/(short)0, acc, /*reuse_a=*/false, /*reuse_b=*/false);
  }

  // D layout: VGPR i -> row = rowTile*16 + i + 8*(lane>>4), col = lane&15
  const int outRowBase = rowTile * 16 + ((lane >> 4) << 3);
  const int outCol = colTile * 16 + (lane & 15);
#pragma unroll
  for (int i = 0; i < 8; ++i)
    C[(long long)(outRowBase + i) * Nout + outCol] = acc[i];
}

// ---------------- edge scatter-aggregate ----------------
// out[dst] += dinv[src]*dinv[dst] * xw[src], C/4 threads per edge, float4 reads.

__global__ void aggregate_kernel(const float* __restrict__ xw,
                                 const long long* __restrict__ src,
                                 const long long* __restrict__ dst,
                                 const float* __restrict__ dinv,
                                 float* __restrict__ out, long long E, int C) {
  long long gid = (long long)blockIdx.x * blockDim.x + threadIdx.x;
  const int tpe = C >> 2;
  long long e = gid / tpe;
  if (e >= E) return;
  int c = (int)(gid % tpe) * 4;
  long long s = src[e], d = dst[e];
  float nrm = dinv[s] * dinv[d];
  const float4 v = *(const float4*)(xw + s * (long long)C + c);
  float* o = out + d * (long long)C + c;
  atomAddF(o + 0, v.x * nrm);
  atomAddF(o + 1, v.y * nrm);
  atomAddF(o + 2, v.z * nrm);
  atomAddF(o + 3, v.w * nrm);
}

// out = [leakyrelu]( agg + dinv^2 * xw + bias )
__global__ void finalize_kernel(float* __restrict__ out,
                                const float* __restrict__ xw,
                                const float* __restrict__ dinv,
                                const float* __restrict__ bias, int C,
                                float slope, int do_relu, long long total) {
  long long i = (long long)blockIdx.x * blockDim.x + threadIdx.x;
  if (i >= total) return;
  long long node = i / C;
  int c = (int)(i % C);
  float di = dinv[node];
  float v = out[i] + di * di * xw[i] + bias[c];
  if (do_relu && v < 0.0f) v *= slope;
  out[i] = v;
}

// ---------------- quantization ----------------
// One 128-thread block per node. Forward value of q == hard selection, so
// pred = max_j Cent[off_j + argmax, :]. Softmax probs only feed `balance`.

__global__ void __launch_bounds__(128)
quantize_kernel(const float* __restrict__ H, const float* __restrict__ Cent,
                float* __restrict__ Hq, float* __restrict__ balance) {
  __shared__ float sH[HID_F];
  __shared__ float sP[480];   // prods[120][4]
  __shared__ float sMax[16];  // [j][d]
  __shared__ float sSum[16];
  __shared__ int   sIdx[16];
  const int t = threadIdx.x;
  const long long node = blockIdx.x;

  sH[t]       = H[node * HID_F + t];
  sH[t + 128] = H[node * HID_F + t + 128];
  __syncthreads();

  if (t < 120) {
#pragma unroll
    for (int d = 0; d < 4; ++d) {
      const float* cptr = Cent + (long long)t * HID_F + d * 64;
      const float* hptr = sH + d * 64;
      float acc = 0.0f;
      for (int s = 0; s < 64; ++s) acc += hptr[s] * cptr[s];
      sP[t * 4 + d] = acc; // TAU == 1.0
    }
  }
  __syncthreads();

  if (t < 16) {
    const int j = t >> 2, d = t & 3;
    const int off = (j == 0) ? 0 : (j == 1) ? 8 : (j == 2) ? 24 : 56;
    const int m = 8 << j;
    float mx = -3.4e38f; int bi = 0;
    for (int k = 0; k < m; ++k) {
      float p = sP[(off + k) * 4 + d];
      if (p > mx) { mx = p; bi = k; }
    }
    float se = 0.0f;
    for (int k = 0; k < m; ++k) se += __expf(sP[(off + k) * 4 + d] - mx);
    sMax[t] = mx; sSum[t] = se; sIdx[t] = bi;
  }
  __syncthreads();

  for (int i = t; i < 480; i += 128) {
    int k = i >> 2, d = i & 3;
    int j = (k < 8) ? 0 : (k < 24) ? 1 : (k < 56) ? 2 : 3;
    float p = __expf(sP[i] - sMax[j * 4 + d]) / sSum[j * 4 + d];
    atomAddF(&balance[i], p);
  }

  for (int c = t; c < HID_F; c += 128) {
    int d = c >> 6;
    float best = -3.4e38f;
#pragma unroll
    for (int j = 0; j < 4; ++j) {
      const int off = (j == 0) ? 0 : (j == 1) ? 8 : (j == 2) ? 24 : 56;
      float v = Cent[(long long)(off + sIdx[j * 4 + d]) * HID_F + c];
      best = fmaxf(best, v);
    }
    Hq[node * HID_F + c] = sH[c] + best;
  }
}

__global__ void reg_kernel(const float* __restrict__ bal, float invN,
                           float* __restrict__ out) {
  float ss = 0.0f;
  for (int k = 0; k < 120; ++k) {
    int j = (k < 8) ? 0 : (k < 24) ? 1 : (k < 56) ? 2 : 3;
    float target = 1.0f / (float)(8 << j);
    for (int d = 0; d < 4; ++d) {
      float v = bal[k * 4 + d] * invN - target;
      ss += v * v;
    }
  }
  *out = sqrtf(ss);
}

// ---------------- host launcher ----------------

extern "C" void kernel_launch(void* const* d_in, const int* in_sizes, int n_in,
                              void* d_out, int out_size, void* d_ws, size_t ws_size,
                              hipStream_t stream) {
  const float*     x    = (const float*)d_in[0];
  const long long* ei   = (const long long*)d_in[1];
  const float*     W0   = (const float*)d_in[2];
  const float*     b0   = (const float*)d_in[3];
  const float*     W1   = (const float*)d_in[4];
  const float*     b1   = (const float*)d_in[5];
  const float*     W2   = (const float*)d_in[6];
  const float*     b2   = (const float*)d_in[7];
  const float*     cent = (const float*)d_in[8];

  const long long N = in_sizes[0] / IN_F;
  const long long E = in_sizes[1] / 2;
  const long long* src = ei;
  const long long* dst = ei + E;
  float* out = (float*)d_out;
  (void)n_in; (void)out_size; (void)ws_size;

  // workspace carve-out (256B aligned)
  char* ws = (char*)d_ws;
  size_t off = 0;
  auto take = [&](size_t bytes) {
    char* p = ws + off;
    off = (off + bytes + 255) & ~(size_t)255;
    return p;
  };
  float*  dinv = (float*)take((size_t)N * 4);
  float*  xw   = (float*)take((size_t)N * HID_F * 4);  // GEMM out; later Hq
  float*  h    = (float*)take((size_t)N * HID_F * 4);  // activations; later N x OUT
  bf16_t* bfA  = (bf16_t*)take((size_t)N * HID_F * 2);
  bf16_t* bfW  = (bf16_t*)take((size_t)HID_F * HID_F * 2);
  float*  bal  = (float*)take(480 * 4);

  const int TB = 256;
  auto nb = [](long long n, int tb) { return (unsigned)((n + tb - 1) / tb); };

  // degree -> dinv (shared by all 3 layers)
  fill_f32<<<nb(N, TB), TB, 0, stream>>>(dinv, 1.0f, N);  // self loops
  deg_kernel<<<nb(E, TB), TB, 0, stream>>>(dst, E, dinv);
  rsqrt_kernel<<<nb(N, TB), TB, 0, stream>>>(dinv, N);

  // ---- layer 1: x @ W0, aggregate, leakyrelu ----
  cvt_bf16_kernel<<<nb(N * IN_F, TB), TB, 0, stream>>>(x, bfA, N * IN_F);
  transpose_cvt_kernel<<<nb(IN_F * HID_F, TB), TB, 0, stream>>>(W0, bfW, IN_F, HID_F);
  {
    dim3 g((unsigned)(N / 16), HID_F / 64);
    gemm_wmma_bf16<<<g, 128, 0, stream>>>(bfA, bfW, xw, IN_F, HID_F);
  }
  fill_f32<<<nb(N * HID_F, TB), TB, 0, stream>>>(h, 0.0f, N * HID_F);
  aggregate_kernel<<<nb(E * (HID_F / 4), TB), TB, 0, stream>>>(xw, src, dst, dinv, h, E, HID_F);
  finalize_kernel<<<nb(N * HID_F, TB), TB, 0, stream>>>(h, xw, dinv, b0, HID_F, 0.5f, 1, N * HID_F);

  // ---- layer 2: h @ W1, aggregate, leakyrelu ----
  cvt_bf16_kernel<<<nb(N * HID_F, TB), TB, 0, stream>>>(h, bfA, N * HID_F);
  transpose_cvt_kernel<<<nb(HID_F * HID_F, TB), TB, 0, stream>>>(W1, bfW, HID_F, HID_F);
  {
    dim3 g((unsigned)(N / 16), HID_F / 64);
    gemm_wmma_bf16<<<g, 128, 0, stream>>>(bfA, bfW, xw, HID_F, HID_F);
  }
  fill_f32<<<nb(N * HID_F, TB), TB, 0, stream>>>(h, 0.0f, N * HID_F);
  aggregate_kernel<<<nb(E * (HID_F / 4), TB), TB, 0, stream>>>(xw, src, dst, dinv, h, E, HID_F);
  finalize_kernel<<<nb(N * HID_F, TB), TB, 0, stream>>>(h, xw, dinv, b1, HID_F, 0.5f, 1, N * HID_F);

  // ---- quantize: hq = h + pred (into xw buffer), balance accumulation ----
  fill_f32<<<nb(480, 64), 64, 0, stream>>>(bal, 0.0f, 480);
  quantize_kernel<<<(unsigned)N, 128, 0, stream>>>(h, cent, xw, bal);

  // ---- layer 3: hq @ W2, aggregate into d_out, bias (no relu) ----
  cvt_bf16_kernel<<<nb(N * HID_F, TB), TB, 0, stream>>>(xw, bfA, N * HID_F);
  transpose_cvt_kernel<<<nb(HID_F * OUT_F, TB), TB, 0, stream>>>(W2, bfW, HID_F, OUT_F);
  {
    dim3 g((unsigned)(N / 16), OUT_F / 64);
    gemm_wmma_bf16<<<g, 128, 0, stream>>>(bfA, bfW, h, HID_F, OUT_F); // h reused as N x OUT
  }
  fill_f32<<<nb(N * OUT_F, TB), TB, 0, stream>>>(out, 0.0f, N * OUT_F);
  aggregate_kernel<<<nb(E * (OUT_F / 4), TB), TB, 0, stream>>>(h, src, dst, dinv, out, E, OUT_F);
  finalize_kernel<<<nb(N * OUT_F, TB), TB, 0, stream>>>(out, h, dinv, b2, OUT_F, 0.5f, 0, N * OUT_F);

  // ---- reg scalar ----
  reg_kernel<<<1, 1, 0, stream>>>(bal, 1.0f / (float)N, out + N * OUT_F);
}
